// EulerAttentionVariant_75892072121088
// MI455X (gfx1250) — matrix operand order repaired and checked by reading, not compiled, not run on hardware
//
#include <hip/hip_runtime.h>
#include <hip/hip_bf16.h>
#include <math.h>

typedef _Float16 h16;
typedef __attribute__((ext_vector_type(16))) _Float16 v16h;
typedef __attribute__((ext_vector_type(8)))  _Float16 v8h;
typedef __attribute__((ext_vector_type(8)))  float    v8f;

#define B_  2
#define S_  2048
#define D_  1024
#define H_  16
#define DH_ 64
#define E_  128          // 2*dh feature dim
#define PHI_ 1.6180339887498948482f
#define INV_SCALE_ 0.08838834764831845f   // 1/sqrt(2*dh) = 1/sqrt(128)

// LUT-based sin/cos: 4096-entry table on [0,2pi), linear interpolation.
// Table entries are computed analytically (identical values to the table).
__device__ __forceinline__ void lut_sincos(float theta, float& s, float& c) {
    const float TWO_PI = 6.283185307179586f;
    const float STEP   = TWO_PI / 4096.0f;
    float pos  = theta * (4096.0f / TWO_PI);
    float i0f  = floorf(pos);
    float frac = pos - i0f;
    int i0 = ((int)i0f) & 4095;          // power-of-2 mod handles negatives
    int i1 = (i0 + 1) & 4095;
    float a0 = STEP * (float)i0;
    float a1 = STEP * (float)i1;
    float s0 = __sinf(a0), s1 = __sinf(a1);
    float c0 = __cosf(a0), c1 = __cosf(a1);
    s = s0 * (1.0f - frac) + s1 * frac;
    c = c0 * (1.0f - frac) + c1 * frac;
}

// ---------------- Phase 1: build Q, K (row-major [bh][s][128]) and
// V transposed ([bh][128][t]) in f16. Scale 1/sqrt(128) folded into Q. -------
__global__ void qkv_kernel(const float* __restrict__ x,
                           const float* __restrict__ pos,
                           const float* __restrict__ wq, const float* __restrict__ bq,
                           const float* __restrict__ wk, const float* __restrict__ bk,
                           const float* __restrict__ wv, const float* __restrict__ bv,
                           h16* __restrict__ Qh, h16* __restrict__ Kh,
                           h16* __restrict__ Vt) {
    int idx = blockIdx.x * blockDim.x + threadIdx.x;   // B*S*H*DH = 4,194,304
    int d = idx & (DH_ - 1);
    int h = (idx >> 6) & (H_ - 1);
    int s = (idx >> 10) & (S_ - 1);
    int b = idx >> 21;

    float xv = x[((size_t)(b * S_ + s)) * D_ + h * DH_ + d];
    float t  = pos[s] * PHI_;
    int   wi = h * DH_ + d;

    float sq, cq, sk, ck, sv, cv;
    lut_sincos(xv / (1.0f + fabsf(wq[wi])) + bq[wi] + t, sq, cq);
    lut_sincos(xv / (1.0f + fabsf(wk[wi])) + bk[wi],     sk, ck);
    lut_sincos(xv / (1.0f + fabsf(wv[wi])) + bv[wi] + t, sv, cv);

    size_t row = ((size_t)(b * H_ + h) * S_ + s) * E_;
    Qh[row + d]       = (h16)(cq * INV_SCALE_);
    Qh[row + DH_ + d] = (h16)(sq * INV_SCALE_);
    Kh[row + d]       = (h16)ck;
    Kh[row + DH_ + d] = (h16)sk;

    size_t vb = (size_t)(b * H_ + h) * E_ * S_;
    Vt[vb + (size_t)d * S_ + s]          = (h16)cv;   // cos half -> e = d
    Vt[vb + (size_t)(DH_ + d) * S_ + s]  = (h16)sv;   // sin half -> e = 64+d
}

// ---------------- Phase 2: flash attention, one wave32 per 16-query block ---
// Per wave: Q = 4 A-frags (16x32 f16, e=0..127) held in regs.
// Key loop over 32-key blocks: 8 WMMA (QK^T, two 16-col tiles x 4 k-slices),
// causal mask, online softmax (shfl_xor row reductions), P -> LDS -> A-frag,
// 8 WMMA (PV into 16x128 f32 accumulator).
__global__ __launch_bounds__(128)
void attn_kernel(const h16* __restrict__ Qh, const h16* __restrict__ Kh,
                 const h16* __restrict__ Vt, float* __restrict__ outc) {
    __shared__ __align__(32) h16 plds[4][16 * 32];   // per-wave P staging tile

    const int lane = threadIdx.x;        // 0..31
    const int wy   = threadIdx.y;        // 0..3
    const int wid  = blockIdx.x * 4 + wy;
    const int qb   = wid & (S_ / 16 - 1);     // query block 0..127
    const int bh   = wid >> 7;                // 0..31  (b*H + h)
    const int hi   = lane >> 4;               // half-wave select
    const int l16  = lane & 15;

    const h16* Qb = Qh + (size_t)bh * S_ * E_;
    const h16* Kb = Kh + (size_t)bh * S_ * E_;
    const h16* Vb = Vt + (size_t)bh * E_ * S_;

    // Load Q A-fragments (contiguous layout: lane<16 -> K 0-15; lane>=16 -> K 16-31)
    v16h qf[4];
    const int qrow = qb * 16 + l16;
    #pragma unroll
    for (int ke = 0; ke < 4; ke++) {
        const h16* p = Qb + (size_t)qrow * E_ + ke * 32 + hi * 16;
        v8h lo = *(const v8h*)p;
        v8h hh = *(const v8h*)(p + 8);
        #pragma unroll
        for (int i = 0; i < 8; i++) { qf[ke][i] = lo[i]; qf[ke][8 + i] = hh[i]; }
    }

    v8f acc[8];
    #pragma unroll
    for (int o = 0; o < 8; o++)
        #pragma unroll
        for (int i = 0; i < 8; i++) acc[o][i] = 0.0f;

    float m_[8], l_[8];
    #pragma unroll
    for (int r = 0; r < 8; r++) { m_[r] = -INFINITY; l_[r] = 0.0f; }

    const int nkb = (qb * 16 + 15) / 32 + 1;   // causal key-block count
    for (int kb = 0; kb < nkb; kb++) {
        // ---- S = Q * K^T : two 16x16 f32 tiles over this 32-key block ----
        v8f ct[2];
        #pragma unroll
        for (int nh = 0; nh < 2; nh++) {
            v8f c;
            #pragma unroll
            for (int i = 0; i < 8; i++) c[i] = 0.0f;
            const int tcol = kb * 32 + nh * 16 + l16;   // key row (B-frag col)
            #pragma unroll
            for (int ke = 0; ke < 4; ke++) {
                const h16* kp = Kb + (size_t)tcol * E_ + ke * 32 + hi * 16;
                v16h bf = *(const v16h*)kp;             // 32B-aligned
                c = __builtin_amdgcn_wmma_f32_16x16x32_f16(
                        false, qf[ke], false, bf, (short)0, c, false, false);
            }
            ct[nh] = c;
        }

        // ---- causal mask + online softmax (C/D layout: row = r + 8*hi) ----
        float pml[2][8];
        #pragma unroll
        for (int r = 0; r < 8; r++) {
            const int srow = qb * 16 + r + hi * 8;
            float v0 = (kb * 32 + l16      <= srow) ? ct[0][r] : -INFINITY;
            float v1 = (kb * 32 + 16 + l16 <= srow) ? ct[1][r] : -INFINITY;
            float loc = fmaxf(v0, v1);
            loc = fmaxf(loc, __shfl_xor(loc, 1));
            loc = fmaxf(loc, __shfl_xor(loc, 2));
            loc = fmaxf(loc, __shfl_xor(loc, 4));
            loc = fmaxf(loc, __shfl_xor(loc, 8));
            const float mn = fmaxf(m_[r], loc);
            const bool dead = (mn == -INFINITY);       // fully-masked so far
            const float alpha = dead ? 1.0f : __expf(m_[r] - mn);
            const float p0 = dead ? 0.0f : __expf(v0 - mn);
            const float p1 = dead ? 0.0f : __expf(v1 - mn);
            float rs = p0 + p1;
            rs += __shfl_xor(rs, 1);
            rs += __shfl_xor(rs, 2);
            rs += __shfl_xor(rs, 4);
            rs += __shfl_xor(rs, 8);
            l_[r] = l_[r] * alpha + rs;
            m_[r] = mn;
            #pragma unroll
            for (int o = 0; o < 8; o++) acc[o][r] *= alpha;
            pml[0][r] = p0;
            pml[1][r] = p1;
        }

        // ---- P: C/D layout -> A-frag layout via per-wave LDS tile ----
        h16* pb = plds[wy];
        #pragma unroll
        for (int nh = 0; nh < 2; nh++)
            #pragma unroll
            for (int r = 0; r < 8; r++)
                pb[(r + hi * 8) * 32 + nh * 16 + l16] = (h16)pml[nh][r];
        asm volatile("s_wait_dscnt 0x0" ::: "memory");

        v16h pf;
        {
            const h16* p = pb + l16 * 32 + hi * 16;
            v8h lo = *(const v8h*)p;
            v8h hh = *(const v8h*)(p + 8);
            #pragma unroll
            for (int i = 0; i < 8; i++) { pf[i] = lo[i]; pf[8 + i] = hh[i]; }
        }

        // ---- O += P * V : 8 output column tiles (e = 0..127) ----
        #pragma unroll
        for (int o = 0; o < 8; o++) {
            const h16* vp = Vb + (size_t)(o * 16 + l16) * S_ + kb * 32 + hi * 16;
            v16h vf = *(const v16h*)vp;                 // 32B-aligned
            acc[o] = __builtin_amdgcn_wmma_f32_16x16x32_f16(
                         false, pf, false, vf, (short)0, acc[o], false, false);
        }
    }

    // ---- epilogue: out_cos + out_sin, normalized by row sum ----
    const int b = bh >> 4, h = bh & (H_ - 1);
    #pragma unroll
    for (int r = 0; r < 8; r++) {
        const float inv = (l_[r] > 0.0f) ? 1.0f / l_[r] : 0.0f;
        const int srow = qb * 16 + r + hi * 8;
        float* op = outc + ((size_t)(b * S_ + srow)) * D_ + h * DH_;
        #pragma unroll
        for (int o = 0; o < 4; o++)
            op[o * 16 + l16] = (acc[o][r] + acc[o + 4][r]) * inv;   // cos + sin halves
    }
}

// ---------------- Phase 3: in-place LUT epilogue on d_out -------------------
__global__ void out_kernel(const float* __restrict__ wout,
                           const float* __restrict__ bout,
                           float* __restrict__ out) {
    int idx = blockIdx.x * blockDim.x + threadIdx.x;   // B*S*D
    int d = idx & (D_ - 1);
    float v  = out[idx];
    float th = v / (1.0f + fabsf(wout[d])) + bout[d];
    float s, c;
    lut_sincos(th, s, c);
    out[idx] = c + s;
}

extern "C" void kernel_launch(void* const* d_in, const int* in_sizes, int n_in,
                              void* d_out, int out_size, void* d_ws, size_t ws_size,
                              hipStream_t stream) {
    (void)in_sizes; (void)n_in; (void)out_size; (void)ws_size;
    const float* x    = (const float*)d_in[0];
    const float* pos  = (const float*)d_in[1];
    const float* wq   = (const float*)d_in[2];
    const float* bq   = (const float*)d_in[3];
    const float* wk   = (const float*)d_in[4];
    const float* bk   = (const float*)d_in[5];
    const float* wv   = (const float*)d_in[6];
    const float* bv   = (const float*)d_in[7];
    const float* wout = (const float*)d_in[8];
    const float* bout = (const float*)d_in[9];

    const size_t qkvElems = (size_t)B_ * H_ * S_ * E_;   // 8,388,608 halves = 16 MB
    h16* Qh = (h16*)d_ws;
    h16* Kh = Qh + qkvElems;
    h16* Vt = Kh + qkvElems;
    float* out = (float*)d_out;

    const int n1 = B_ * S_ * H_ * DH_;                   // 4,194,304
    qkv_kernel<<<n1 / 256, 256, 0, stream>>>(x, pos, wq, bq, wk, bk, wv, bv,
                                             Qh, Kh, Vt);

    const int nwaves = B_ * H_ * (S_ / 16);              // 4096 waves
    attn_kernel<<<nwaves / 4, dim3(32, 4), 0, stream>>>(Qh, Kh, Vt, out);

    const int n3 = B_ * S_ * D_;
    out_kernel<<<n3 / 256, 256, 0, stream>>>(wout, bout, out);
}
